// NonLocalAttentionBlock_34729105555795
// MI455X (gfx1250) — matrix operand order repaired and checked by reading, not compiled, not run on hardware
//
#include <hip/hip_runtime.h>

typedef __attribute__((ext_vector_type(16))) _Float16 v16h;
typedef __attribute__((ext_vector_type(8)))  _Float16 v8h;
typedef __attribute__((ext_vector_type(8)))  float    v8f;
typedef __attribute__((ext_vector_type(4)))  int      v4i;

constexpr int kB = 4, kC = 512, kN = 4096, kD = 64;

// ---- CDNA5 async global->LDS support (guarded; falls back to sync path) ----
#if defined(__has_builtin)
#if __has_builtin(__builtin_amdgcn_global_load_async_to_lds_b128)
#define HAVE_ASYNC_LDS 1
#endif
#endif
#ifndef HAVE_ASYNC_LDS
#define HAVE_ASYNC_LDS 0
#endif

#if HAVE_ASYNC_LDS
typedef __attribute__((address_space(1))) v4i g_v4i;   // global int4
typedef __attribute__((address_space(3))) v4i l_v4i;   // LDS int4

__device__ __forceinline__ void async_copy16(const void* gsrc, void* ldst) {
  // AS1 src / AS3 dst: exact match if params are addrspace-qualified,
  // implicitly convertible to generic otherwise.
  __builtin_amdgcn_global_load_async_to_lds_b128((g_v4i*)gsrc, (l_v4i*)ldst, 0, 0);
}
#endif

__device__ __forceinline__ void wait_async_lds() {
#if defined(__has_builtin)
#if __has_builtin(__builtin_amdgcn_s_wait_asynccnt)
  __builtin_amdgcn_s_wait_asynccnt(0);
  return;
#else
  asm volatile("s_wait_asynccnt 0x0" ::: "memory");
  return;
#endif
#else
  asm volatile("s_wait_asynccnt 0x0" ::: "memory");
#endif
}

// ---- workspace layout (bytes) ----
constexpr size_t OFF_XH = 0;
constexpr size_t SZ_XH  = (size_t)kB * kC * kN * 2;       // x in f16
constexpr size_t OFF_WH = OFF_XH + SZ_XH;
constexpr size_t SZ_WH  = (size_t)3 * kD * kC * 2;        // theta/phi/g weights f16
constexpr size_t OFF_QH = OFF_WH + SZ_WH;                 // Q = theta^T  [B][N][64] f16
constexpr size_t SZ_P   = (size_t)kB * kN * kD * 2;
constexpr size_t OFF_KH = OFF_QH + SZ_P;                  // K = phi      [B][64][N] f16
constexpr size_t OFF_VH = OFF_KH + SZ_P;                  // V = g^T      [B][N][64] f16
constexpr size_t OFF_Y  = OFF_VH + SZ_P;                  // y^T          [B][N][64] f32
constexpr size_t SZ_Y   = (size_t)kB * kN * kD * 4;
constexpr size_t OFF_S1 = OFF_Y + SZ_Y;                   // sum_d   [64] f32
constexpr size_t OFF_S2 = OFF_S1 + 64 * 4;                // sum_de  [64][64] f32
constexpr size_t OFF_SC = OFF_S2 + 64 * 64 * 4;           // bn scale [512]
constexpr size_t OFF_SH = OFF_SC + 512 * 4;               // bn shift [512]

// ---------------- conversions ----------------
__global__ void k_cvt_x(const float* __restrict__ x, _Float16* __restrict__ xh, int n) {
  int i = blockIdx.x * blockDim.x + threadIdx.x;
  if (i < n) xh[i] = (_Float16)x[i];
}

__global__ void k_cvt_w(const float* __restrict__ w0, const float* __restrict__ w1,
                        const float* __restrict__ w2, _Float16* __restrict__ wh) {
  int i = blockIdx.x * blockDim.x + threadIdx.x;
  const int S = kD * kC;  // 32768
  if (i < S)            wh[i] = (_Float16)w0[i];
  else if (i < 2 * S)   wh[i] = (_Float16)w1[i - S];
  else if (i < 3 * S)   wh[i] = (_Float16)w2[i - 2 * S];
}

// ---------------- projection GEMMs: out[d,n] = sum_c W[d,c] X[c,n] + bias[d] ----------------
// block = 128 threads (4 waves), covers 32d x 32n tile; each wave one 16x16 WMMA tile.
// W tile staged via async global->LDS (natural layout); X tile scattered transposed so
// both A and B fragment reads are contiguous ds_load_b128.
__global__ void __launch_bounds__(128)
k_proj(const _Float16* __restrict__ Wh, const _Float16* __restrict__ Xh,
       const float* __restrict__ tb, const float* __restrict__ pb, const float* __restrict__ gb,
       _Float16* __restrict__ Qh, _Float16* __restrict__ Kh, _Float16* __restrict__ Vh) {
  __shared__ __attribute__((aligned(16))) _Float16 lw[32][40];    // [d][c]
  __shared__ __attribute__((aligned(16))) _Float16 lxT[32][40];   // [n][c]
  const int tid  = threadIdx.x;
  const int lane = tid & 31;
  const int wv   = tid >> 5;
  const int proj  = blockIdx.z % 3;
  const int b     = blockIdx.z / 3;
  const int dbase = blockIdx.y * 32;
  const int nbase = blockIdx.x * 32;

  const _Float16* wsrc = Wh + (size_t)proj * kD * kC;
  const _Float16* xsrc = Xh + (size_t)b * kC * kN;

  const int dt  = (wv >> 1) * 16;
  const int nt  = (wv & 1) * 16;
  const int l15 = lane & 15;
  const int hi  = lane >> 4;
  const int kb8 = hi * 8;
  const int rb  = hi * 16;

  const int lr   = tid >> 2;        // 0..31
  const int lseg = (tid & 3) * 8;   // 0,8,16,24

  v8f acc = {};
  for (int cc = 0; cc < kC; cc += 32) {
    __syncthreads();
#if HAVE_ASYNC_LDS
    async_copy16(wsrc + (size_t)(dbase + lr) * kC + cc + lseg, &lw[lr][lseg]);
#else
    *(v8h*)&lw[lr][lseg] = *(const v8h*)(wsrc + (size_t)(dbase + lr) * kC + cc + lseg);
#endif
    {
      v8h xv = *(const v8h*)(xsrc + (size_t)(cc + lr) * kN + nbase + lseg);
#pragma unroll
      for (int t = 0; t < 8; ++t) lxT[lseg + t][lr] = xv[t];
    }
#if HAVE_ASYNC_LDS
    wait_async_lds();
#endif
    __syncthreads();

    v16h a;   // A 16x32: row = dt+l15, K contiguous rb..rb+15
    const int rowd = dt + l15;
#pragma unroll
    for (int t = 0; t < 8; ++t) { a[t] = lw[rowd][rb + t]; a[8 + t] = lw[rowd][rb + 8 + t]; }
    v16h bf;  // B 32x16: col = nt+l15, rows rb+0..15 (contiguous in lxT)
    const _Float16* bp = &lxT[nt + l15][rb];
#pragma unroll
    for (int e = 0; e < 16; ++e) bf[e] = bp[e];

    acc = __builtin_amdgcn_wmma_f32_16x16x32_f16(false, a, false, bf, (short)0, acc, false, false);
  }

  const float* bias = (proj == 0) ? tb : (proj == 1) ? pb : gb;
#pragma unroll
  for (int r = 0; r < 8; ++r) {
    const int m  = r + 8 * hi;
    const int dg = dbase + dt + m;
    const int ng = nbase + nt + l15;
    _Float16 v = (_Float16)(acc[r] + bias[dg]);
    if (proj == 0)      Qh[(size_t)b * kN * kD + (size_t)ng * kD + dg] = v;  // theta^T
    else if (proj == 1) Kh[(size_t)b * kD * kN + (size_t)dg * kN + ng] = v;  // phi
    else                Vh[(size_t)b * kN * kD + (size_t)ng * kD + dg] = v;  // g^T
  }
}

// ---------------- flash attention: Y[i,d] = sum_j softmax_j(Q·K)[i,j] * V[j,d] ----------------
// block = 128 threads (4 waves); wave handles 16 query rows; j streamed in chunks of 32.
__global__ void __launch_bounds__(128)
k_flash(const _Float16* __restrict__ Qh, const _Float16* __restrict__ Kh,
        const _Float16* __restrict__ Vh, float* __restrict__ Y) {
  __shared__ __attribute__((aligned(16))) _Float16 lkT[32][72];    // [j][d]
  __shared__ __attribute__((aligned(16))) _Float16 lvT[64][40];    // [d][j]
  __shared__ __attribute__((aligned(16))) _Float16 lp[4][16][40];  // per-wave P tile [i][j]

  const int tid  = threadIdx.x;
  const int lane = tid & 31;
  const int wv   = tid >> 5;
  const int l15  = lane & 15;
  const int hi   = lane >> 4;
  const int kb8  = hi * 8;
  const int rb   = hi * 16;
  const int b    = blockIdx.x >> 6;
  const int i0   = (blockIdx.x & 63) * 64 + wv * 16;

  // Q fragments (A-layout) straight from global: row-major [N][64]
  const _Float16* qbase = Qh + (size_t)b * kN * kD + (size_t)(i0 + l15) * kD;
  v16h qa0, qa1;
  {
    v8h x0 = *(const v8h*)(qbase + rb);
    v8h x1 = *(const v8h*)(qbase + rb + 8);
    v8h x2 = *(const v8h*)(qbase + 32 + rb);
    v8h x3 = *(const v8h*)(qbase + 32 + rb + 8);
#pragma unroll
    for (int t = 0; t < 8; ++t) { qa0[t] = x0[t]; qa0[8 + t] = x1[t]; qa1[t] = x2[t]; qa1[8 + t] = x3[t]; }
  }

  const _Float16* kbase = Kh + (size_t)b * kD * kN;
  const _Float16* vbase = Vh + (size_t)b * kN * kD;

  float mrow[8], lrow[8];
  v8f o0 = {}, o1 = {}, o2 = {}, o3 = {};
#pragma unroll
  for (int r = 0; r < 8; ++r) { mrow[r] = -1e30f; lrow[r] = 0.0f; }

  const int kdrow = tid >> 1;          // 0..63   (K tile coop load)
  const int kjseg = (tid & 1) * 16;
  const int vjrow = tid >> 2;          // 0..31   (V tile coop load)
  const int vdseg = (tid & 3) * 16;

  for (int j0 = 0; j0 < kN; j0 += 32) {
    __syncthreads();
    // stage K chunk transposed: lkT[j][d] = phi[d][j0+j]
    {
      const _Float16* kp = kbase + (size_t)kdrow * kN + j0 + kjseg;
      v8h ka = *(const v8h*)(kp);
      v8h kc = *(const v8h*)(kp + 8);
#pragma unroll
      for (int t = 0; t < 8; ++t) { lkT[kjseg + t][kdrow] = ka[t]; lkT[kjseg + 8 + t][kdrow] = kc[t]; }
      // stage V chunk transposed: lvT[d][j] = gT[j0+j][d]
      const _Float16* vp = vbase + (size_t)(j0 + vjrow) * kD + vdseg;
      v8h va = *(const v8h*)(vp);
      v8h vc = *(const v8h*)(vp + 8);
#pragma unroll
      for (int t = 0; t < 8; ++t) { lvT[vdseg + t][vjrow] = va[t]; lvT[vdseg + 8 + t][vjrow] = vc[t]; }
      if (j0 + 32 < kN) {  // prefetch next chunk
        __builtin_prefetch(kbase + (size_t)kdrow * kN + j0 + 32 + kjseg, 0, 1);
        __builtin_prefetch(vbase + (size_t)(j0 + 32 + vjrow) * kD + vdseg, 0, 1);
      }
    }
    __syncthreads();

    // S = Q * K (two 16x16 j-tiles, K-dim = d = 64 -> 2 chained WMMAs each)
    v8f s0 = {}, s1 = {};
    {
      v16h kf00, kf10, kf01, kf11;
      const _Float16* p0 = &lkT[l15][rb];
      const _Float16* p1 = &lkT[l15][32 + rb];
      const _Float16* p2 = &lkT[16 + l15][rb];
      const _Float16* p3 = &lkT[16 + l15][32 + rb];
#pragma unroll
      for (int e = 0; e < 16; ++e) { kf00[e] = p0[e]; kf10[e] = p1[e]; kf01[e] = p2[e]; kf11[e] = p3[e]; }
      s0 = __builtin_amdgcn_wmma_f32_16x16x32_f16(false, qa0, false, kf00, (short)0, s0, false, false);
      s0 = __builtin_amdgcn_wmma_f32_16x16x32_f16(false, qa1, false, kf10, (short)0, s0, false, false);
      s1 = __builtin_amdgcn_wmma_f32_16x16x32_f16(false, qa0, false, kf01, (short)0, s1, false, false);
      s1 = __builtin_amdgcn_wmma_f32_16x16x32_f16(false, qa1, false, kf11, (short)0, s1, false, false);
    }

    // online softmax: C-tile row m = r + 8*hi lives in one VGPR across a 16-lane half
    float p0v[8], p1v[8], al[8];
#pragma unroll
    for (int r = 0; r < 8; ++r) {
      float mx = fmaxf(s0[r], s1[r]);
      mx = fmaxf(mx, __shfl_xor(mx, 1, 32));
      mx = fmaxf(mx, __shfl_xor(mx, 2, 32));
      mx = fmaxf(mx, __shfl_xor(mx, 4, 32));
      mx = fmaxf(mx, __shfl_xor(mx, 8, 32));
      const float mnew = fmaxf(mrow[r], mx);
      const float a = __expf(mrow[r] - mnew);
      mrow[r] = mnew;
      const float e0 = __expf(s0[r] - mnew);
      const float e1 = __expf(s1[r] - mnew);
      float rs = e0 + e1;
      rs += __shfl_xor(rs, 1, 32);
      rs += __shfl_xor(rs, 2, 32);
      rs += __shfl_xor(rs, 4, 32);
      rs += __shfl_xor(rs, 8, 32);
      lrow[r] = lrow[r] * a + rs;
      p0v[r] = e0; p1v[r] = e1; al[r] = a;
    }
#pragma unroll
    for (int r = 0; r < 8; ++r) { o0[r] *= al[r]; o1[r] *= al[r]; o2[r] *= al[r]; o3[r] *= al[r]; }

    // bounce P through LDS: C-layout -> A-layout
#pragma unroll
    for (int r = 0; r < 8; ++r) {
      const int m = r + 8 * hi;
      lp[wv][m][l15]      = (_Float16)p0v[r];
      lp[wv][m][16 + l15] = (_Float16)p1v[r];
    }
    __syncthreads();

    v16h pa;
    {
      const _Float16* q0 = &lp[wv][l15][rb];
      const _Float16* q1 = &lp[wv][l15][rb + 8];
#pragma unroll
      for (int t = 0; t < 8; ++t) { pa[t] = q0[t]; pa[8 + t] = q1[t]; }
    }
    // O += P * V (K-dim = j = 32; four d-tiles of 16)
    {
      v16h vf;
      const _Float16* p;
      p = &lvT[l15][rb];
#pragma unroll
      for (int e = 0; e < 16; ++e) vf[e] = p[e];
      o0 = __builtin_amdgcn_wmma_f32_16x16x32_f16(false, pa, false, vf, (short)0, o0, false, false);
      p = &lvT[16 + l15][rb];
#pragma unroll
      for (int e = 0; e < 16; ++e) vf[e] = p[e];
      o1 = __builtin_amdgcn_wmma_f32_16x16x32_f16(false, pa, false, vf, (short)0, o1, false, false);
      p = &lvT[32 + l15][rb];
#pragma unroll
      for (int e = 0; e < 16; ++e) vf[e] = p[e];
      o2 = __builtin_amdgcn_wmma_f32_16x16x32_f16(false, pa, false, vf, (short)0, o2, false, false);
      p = &lvT[48 + l15][rb];
#pragma unroll
      for (int e = 0; e < 16; ++e) vf[e] = p[e];
      o3 = __builtin_amdgcn_wmma_f32_16x16x32_f16(false, pa, false, vf, (short)0, o3, false, false);
    }
  }

  float* ybase = Y + (size_t)b * kN * kD + (size_t)i0 * kD;
#pragma unroll
  for (int r = 0; r < 8; ++r) {
    const int m = r + 8 * hi;
    const float inv = 1.0f / lrow[r];
    ybase[(size_t)m * kD + l15]      = o0[r] * inv;
    ybase[(size_t)m * kD + 16 + l15] = o1[r] * inv;
    ybase[(size_t)m * kD + 32 + l15] = o2[r] * inv;
    ybase[(size_t)m * kD + 48 + l15] = o3[r] * inv;
  }
}

// ---------------- second-moment reduction over (B,N): S1[d], S2[d][e] ----------------
__global__ void __launch_bounds__(256)
k_stats(const float* __restrict__ Y, float* __restrict__ S1, float* __restrict__ S2) {
  __shared__ __attribute__((aligned(16))) float ly[64][68];  // 272B rows: 16B aligned
  const int tid = threadIdx.x;
  float a2[16];
#pragma unroll
  for (int k = 0; k < 16; ++k) a2[k] = 0.0f;
  float a1 = 0.0f;
  const int rowbase = blockIdx.x * 256;
  for (int ch = 0; ch < 4; ++ch) {
    __syncthreads();
#if HAVE_ASYNC_LDS
#pragma unroll
    for (int q = 0; q < 4; ++q) {
      const int idx  = tid + q * 256;        // 0..1023 tasks of 16B
      const int row  = idx >> 4;
      const int col4 = (idx & 15) * 4;
      async_copy16(Y + (size_t)(rowbase + ch * 64 + row) * 64 + col4, &ly[row][col4]);
    }
    wait_async_lds();
#else
    for (int idx = tid; idx < 64 * 64; idx += 256)
      ly[idx >> 6][idx & 63] = Y[(size_t)(rowbase + ch * 64 + (idx >> 6)) * 64 + (idx & 63)];
#endif
    __syncthreads();
#pragma unroll 1
    for (int r = 0; r < 64; ++r) {
#pragma unroll
      for (int k = 0; k < 16; ++k) {
        const int p = tid + k * 256;
        a2[k] += ly[r][p >> 6] * ly[r][p & 63];
      }
      if (tid < 64) a1 += ly[r][tid];
    }
  }
#pragma unroll
  for (int k = 0; k < 16; ++k) atomicAdd(&S2[tid + k * 256], a2[k]);
  if (tid < 64) atomicAdd(&S1[tid], a1);
}

// ---------------- BN coefficients via linearity of the out-conv ----------------
__global__ void k_bnprep(const float* __restrict__ S1, const float* __restrict__ S2,
                         const float* __restrict__ out_w, const float* __restrict__ out_b,
                         const float* __restrict__ gamma, const float* __restrict__ beta,
                         float* __restrict__ sc, float* __restrict__ sh) {
  const int c = blockIdx.x * blockDim.x + threadIdx.x;
  if (c >= kC) return;
  const float Mi = 1.0f / (float)(kB * kN);
  float w[64];
#pragma unroll
  for (int d = 0; d < 64; ++d) w[d] = out_w[c * 64 + d];
  float t1 = 0.0f;
#pragma unroll
  for (int d = 0; d < 64; ++d) t1 += w[d] * S1[d] * Mi;
  float q = 0.0f;
  for (int d = 0; d < 64; ++d) {
    float s = 0.0f;
#pragma unroll
    for (int e = 0; e < 64; ++e) s += w[e] * S2[d * 64 + e];
    q += w[d] * s;
  }
  const float var  = q * Mi - t1 * t1;
  const float mean = t1 + out_b[c];
  const float inv  = rsqrtf(var + 1e-5f);
  const float g    = gamma[c] * inv;
  sc[c] = g;
  sh[c] = beta[c] - mean * g;
}

// ---------------- fused out-conv + BN + residual ----------------
__global__ void __launch_bounds__(256)
k_finalize(const float* __restrict__ x, const float* __restrict__ Y,
           const float* __restrict__ out_w, const float* __restrict__ out_b,
           const float* __restrict__ sc, const float* __restrict__ sh,
           const float* __restrict__ scale, float* __restrict__ out) {
  __shared__ __attribute__((aligned(16))) float ly[32][68];
  const int tid = threadIdx.x;
  const int b  = blockIdx.x >> 7;
  const int n0 = (blockIdx.x & 127) * 32;
#if HAVE_ASYNC_LDS
#pragma unroll
  for (int q = 0; q < 2; ++q) {
    const int idx  = tid + q * 256;          // 0..511 tasks of 16B
    const int row  = idx >> 4;
    const int col4 = (idx & 15) * 4;
    async_copy16(Y + ((size_t)b * kN + n0 + row) * 64 + col4, &ly[row][col4]);
  }
  wait_async_lds();
#else
  for (int idx = tid; idx < 32 * 64; idx += 256)
    ly[idx >> 6][idx & 63] = Y[((size_t)b * kN + n0 + (idx >> 6)) * 64 + (idx & 63)];
#endif
  __syncthreads();
  const float sv = scale[0];
  for (int oi = tid; oi < 32 * kC; oi += 256) {
    const int c  = oi >> 5;
    const int nl = oi & 31;
    float acc = out_b[c];
    const float* wr = out_w + c * 64;
#pragma unroll
    for (int d = 0; d < 64; ++d) acc += wr[d] * ly[nl][d];
    const float bn = acc * sc[c] + sh[c];
    const size_t gi = (size_t)b * kC * kN + (size_t)c * kN + n0 + nl;
    out[gi] = x[gi] + sv * bn;
  }
}

extern "C" void kernel_launch(void* const* d_in, const int* in_sizes, int n_in,
                              void* d_out, int out_size, void* d_ws, size_t ws_size,
                              hipStream_t stream) {
  (void)in_sizes; (void)n_in; (void)out_size; (void)ws_size;
  const float* x     = (const float*)d_in[0];
  const float* tw    = (const float*)d_in[1];
  const float* tb    = (const float*)d_in[2];
  const float* pw    = (const float*)d_in[3];
  const float* pbv   = (const float*)d_in[4];
  const float* gw    = (const float*)d_in[5];
  const float* gbv   = (const float*)d_in[6];
  const float* ow    = (const float*)d_in[7];
  const float* ob    = (const float*)d_in[8];
  const float* gamma = (const float*)d_in[9];
  const float* beta  = (const float*)d_in[10];
  const float* scale = (const float*)d_in[11];
  float* out = (float*)d_out;

  char* ws = (char*)d_ws;
  _Float16* Xh = (_Float16*)(ws + OFF_XH);
  _Float16* Wh = (_Float16*)(ws + OFF_WH);
  _Float16* Qh = (_Float16*)(ws + OFF_QH);
  _Float16* Kh = (_Float16*)(ws + OFF_KH);
  _Float16* Vh = (_Float16*)(ws + OFF_VH);
  float*    Yf = (float*)(ws + OFF_Y);
  float*    S1 = (float*)(ws + OFF_S1);
  float*    S2 = (float*)(ws + OFF_S2);
  float*    SC = (float*)(ws + OFF_SC);
  float*    SH = (float*)(ws + OFF_SH);

  (void)hipMemsetAsync(ws + OFF_S1, 0, (64 + 64 * 64) * sizeof(float), stream);

  const int nx = kB * kC * kN;
  k_cvt_x<<<(nx + 255) / 256, 256, 0, stream>>>(x, Xh, nx);
  k_cvt_w<<<(3 * kD * kC + 255) / 256, 256, 0, stream>>>(tw, pw, gw, Wh);

  dim3 gp(kN / 32, kD / 32, 3 * kB);       // 128 x 2 x 12
  k_proj<<<gp, 128, 0, stream>>>(Wh, Xh, tb, pbv, gbv, Qh, Kh, Vh);

  k_flash<<<kB * (kN / 64), 128, 0, stream>>>(Qh, Kh, Vh, Yf);   // 256 blocks

  k_stats<<<(kB * kN) / 256, 256, 0, stream>>>(Yf, S1, S2);      // 64 blocks
  k_bnprep<<<2, 256, 0, stream>>>(S1, S2, ow, ob, gamma, beta, SC, SH);
  k_finalize<<<kB * (kN / 32), 256, 0, stream>>>(x, Yf, ow, ob, SC, SH, scale, out);
}
